// GritMessagePassingLayer_24824910970956
// MI455X (gfx1250) — matrix operand order repaired and checked by reading, not compiled, not run on hardware
//
#include <hip/hip_runtime.h>
#include <hip/hip_bf16.h>
#include <math.h>

typedef __attribute__((ext_vector_type(16))) _Float16 v16h;
typedef __attribute__((ext_vector_type(8)))  _Float16 h8;
typedef __attribute__((ext_vector_type(8)))  float    v8f;
typedef __attribute__((ext_vector_type(4)))  float    f32x4;

#define HID   256
#define NHEAD 8
#define DHEAD 32
#define GF_BIAS 1
#define GF_RES  2
#define GF_RELU 4

#define ASTR 40   // padded LDS row stride (halfs): 80B -> 16B-aligned frag reads, conflict-free banks
#define BSTR 40

union HFrag { v16h v; h8 h[2]; };

// ---------------------------------------------------------------- cvt f32->f16
__global__ __launch_bounds__(256) void grit_cvt_f16(const float* __restrict__ s,
                                                    _Float16* __restrict__ d, int n) {
  int i = blockIdx.x * 256 + threadIdx.x;
  if (i < n) d[i] = (_Float16)s[i];
}

// ---------------------------------------------------------------- WMMA GEMM
// C[M,N] = A[M,K](f32) @ B[K,N](f16)  [+bias] [+res] [relu]
// block: 256 thr = 8 waves; block tile 64x64; wave tile 16x32 (two 16x16 frags)
// A staged row-major [m][k]; B staged TRANSPOSED [n][k] so each lane's fragment
// is 16 contiguous halfs -> two ds_load_b128 per fragment (no scalar gathers).
__global__ __launch_bounds__(256) void grit_gemm(
    const float* __restrict__ A, const _Float16* __restrict__ B,
    const float* __restrict__ bias, const float* res,
    float* C, int M, int K, int N, int flags)
{
  __shared__ _Float16 As[64 * ASTR];
  __shared__ _Float16 Bst[64 * BSTR];
  const int tid  = threadIdx.x;
  const int lane = tid & 31;
  const int wave = tid >> 5;
  const int wm   = wave & 3;       // 0..3  : 16-row tile
  const int wn   = wave >> 2;      // 0..1  : 32-col tile
  const int row0 = blockIdx.x * 64;
  const int col0 = blockIdx.y * 64;

  v8f c0 = {}; v8f c1 = {};

  const int ar = tid >> 2;            // 0..63 staging row of A
  const int ac = (tid & 3) * 8;       // 0..24 staging col of A
  const int br = tid >> 3;            // 0..31 staging k-row of B
  const int bc = (tid & 7) * 8;       // 0..56 staging n-col of B

  // fragment lane mapping (row/col by lane%16; k contiguous: k = 16*(lane/16) + s)
  const int mrow  = wm * 16 + (lane & 15);
  const int kbA   = (lane >> 4) * 16;    // A: halfs 0..15 -> K=kbA..kbA+15
  const int kbB   = (lane >> 4) * 16;    // B: halfs 0..15 -> K=kbB..kbB+15
  const int ncol  = wn * 32 + (lane & 15);

  const _Float16* aF0 = &As[mrow * ASTR + kbA];
  const _Float16* aF1 = &As[mrow * ASTR + 8 + kbA];
  const _Float16* bF0 = &Bst[ncol * BSTR + kbB];
  const _Float16* bF1 = &Bst[(ncol + 16) * BSTR + kbB];

  for (int kt = 0; kt < K; kt += 32) {
    { // stage A tile 64x32, convert f32 -> f16, vector LDS store
      const int grow = row0 + ar;
      h8 hv;
      if (grow < M) {
        const float* ap = A + (size_t)grow * K + kt + ac;
        f32x4 v0 = *(const f32x4*)ap;
        f32x4 v1 = *(const f32x4*)(ap + 4);
        #pragma unroll
        for (int j = 0; j < 4; ++j) { hv[j] = (_Float16)v0[j]; hv[4 + j] = (_Float16)v1[j]; }
      } else {
        #pragma unroll
        for (int j = 0; j < 8; ++j) hv[j] = (_Float16)0.0f;
      }
      *(h8*)&As[ar * ASTR + ac] = hv;
    }
    { // stage B tile 32x64 transposed: global b128 read, scattered b16 stores
      const _Float16* bp = B + (size_t)(kt + br) * N + col0 + bc;
      h8 bq = *(const h8*)bp;
      #pragma unroll
      for (int j = 0; j < 8; ++j) Bst[(bc + j) * BSTR + br] = bq[j];
    }
    __syncthreads();

    HFrag a, b0, b1;
    a.h[0]  = *(const h8*)aF0;
    a.h[1]  = *(const h8*)aF1;
    b0.h[0] = *(const h8*)bF0;
    b0.h[1] = *(const h8*)(bF0 + 8);
    b1.h[0] = *(const h8*)bF1;
    b1.h[1] = *(const h8*)(bF1 + 8);

    c0 = __builtin_amdgcn_wmma_f32_16x16x32_f16(false, a.v, false, b0.v, (short)0, c0, false, false);
    c1 = __builtin_amdgcn_wmma_f32_16x16x32_f16(false, a.v, false, b1.v, (short)0, c1, false, false);
    __syncthreads();
  }

  // epilogue: C f32 layout (lane n = L%16; VGPR v -> m = v + (L>=16 ? 8:0))
  const int colA  = col0 + wn * 32 + (lane & 15);
  const int colB  = colA + 16;
  const int rbase = row0 + wm * 16 + ((lane >> 4) << 3);
  #pragma unroll
  for (int v = 0; v < 8; ++v) {
    int r = rbase + v;
    if (r < M) {
      float o0 = c0[v], o1 = c1[v];
      if (flags & GF_BIAS) { o0 += bias[colA]; o1 += bias[colB]; }
      if (flags & GF_RES)  { o0 += res[(size_t)r * N + colA]; o1 += res[(size_t)r * N + colB]; }
      if (flags & GF_RELU) { o0 = fmaxf(o0, 0.0f); o1 = fmaxf(o1, 0.0f); }
      C[(size_t)r * N + colA] = o0;
      C[(size_t)r * N + colB] = o1;
    }
  }
}

// ------------------------------------------------- edge message (signed sqrt)
// cA may alias Ew (in-place): per-element read-then-write, no restrict on those
__global__ __launch_bounds__(256) void grit_edge_msg(
    const float* __restrict__ Qh, const float* __restrict__ Kh,
    const float* Ew, const float* __restrict__ Eb,
    const int* __restrict__ ei, int E, float* cA)
{
  const int e = blockIdx.x;
  const int c = threadIdx.x;
  const int dst = ei[e], src = ei[E + e];
  const size_t i = (size_t)e * HID + c;
  float c1 = (Qh[(size_t)dst * HID + c] + Kh[(size_t)src * HID + c]) * Ew[i];
  float c2 = (c1 >= 0.0f) ? sqrtf(c1) : -sqrtf(-c1);
  cA[i] = fmaxf(c2 + Eb[i], 0.0f);
}

// ---------------------------------------- per-head score + softmax denominator
__global__ __launch_bounds__(256) void grit_score(
    const float* __restrict__ Oe, const float* __restrict__ Aw,
    const int* __restrict__ ei, int E,
    float* __restrict__ score, float* denom)
{
  __shared__ float AwS[DHEAD * NHEAD];
  const int tid = threadIdx.x;
  AwS[tid] = Aw[tid];
  __syncthreads();
  int i = blockIdx.x * 256 + tid;
  if (i < E * NHEAD) {
    int e = i >> 3, h = i & 7;
    const float* oe = Oe + (size_t)e * HID + h * DHEAD;
    float s = 0.0f;
    #pragma unroll
    for (int d = 0; d < DHEAD; ++d) s += oe[d] * AwS[d * NHEAD + h];
    s = fminf(fmaxf(s, -5.0f), 5.0f);
    score[i] = s;
    atomicAdd(&denom[ei[e] * NHEAD + h], __expf(s));
  }
}

// --------------------------------------------------- weighted scatter to nodes
__global__ __launch_bounds__(256) void grit_scatter(
    const float* __restrict__ score, const float* __restrict__ denom,
    const float* __restrict__ Vh, const float* __restrict__ Oe,
    const int* __restrict__ ei, int E, float* agg, float* rowR)
{
  const int e = blockIdx.x;
  const int c = threadIdx.x;
  const int h = c >> 5;
  const int dst = ei[e], src = ei[E + e];
  float w = __expf(score[(size_t)e * NHEAD + h]) / (denom[dst * NHEAD + h] + 1e-16f);
  atomicAdd(&agg [(size_t)dst * HID + c], Vh[(size_t)src * HID + c] * w);
  atomicAdd(&rowR[(size_t)dst * HID + c], Oe[(size_t)e   * HID + c] * w);
}

// -------------------------------------------- On = agg + einsum(rowR, BW) in place
__global__ __launch_bounds__(256) void grit_bw(
    const float* __restrict__ rowR, const float* __restrict__ BW, float* agg)
{
  __shared__ float r[HID];
  const int tid = threadIdx.x;
  const size_t base = (size_t)blockIdx.x * HID;
  r[tid] = rowR[base + tid];
  __syncthreads();
  const int h = tid >> 5;
  float acc = agg[base + tid];
  #pragma unroll
  for (int d = 0; d < DHEAD; ++d) acc += r[h * DHEAD + d] * BW[d * HID + tid];
  agg[base + tid] = acc;
}

// ------------------------------------------------------- in-place row LayerNorm
__global__ __launch_bounds__(256) void grit_ln(float* X, const float* __restrict__ g,
                                               const float* __restrict__ b)
{
  __shared__ float red[256];
  const int tid = threadIdx.x;
  const size_t base = (size_t)blockIdx.x * HID;
  float x = X[base + tid];
  red[tid] = x; __syncthreads();
  for (int s = 128; s > 0; s >>= 1) { if (tid < s) red[tid] += red[tid + s]; __syncthreads(); }
  float mean = red[0] * (1.0f / HID);
  __syncthreads();
  float d = x - mean;
  red[tid] = d * d; __syncthreads();
  for (int s = 128; s > 0; s >>= 1) { if (tid < s) red[tid] += red[tid + s]; __syncthreads(); }
  float var = red[0] * (1.0f / HID);
  X[base + tid] = d * rsqrtf(var + 1e-5f) * g[tid] + b[tid];
}

// =============================================================== host launcher
extern "C" void kernel_launch(void* const* d_in, const int* in_sizes, int n_in,
                              void* d_out, int out_size, void* d_ws, size_t ws_size,
                              hipStream_t stream) {
  (void)n_in; (void)out_size; (void)ws_size;
  const float* x    = (const float*)d_in[0];
  const float* conn = (const float*)d_in[1];
  const float* Wq   = (const float*)d_in[2];
  const float* Wk   = (const float*)d_in[3];
  const float* Wv   = (const float*)d_in[4];
  const float* Wew  = (const float*)d_in[5];
  const float* Web  = (const float*)d_in[6];
  const float* beb  = (const float*)d_in[7];
  const float* Weo  = (const float*)d_in[8];
  const float* beo  = (const float*)d_in[9];
  const float* Aw   = (const float*)d_in[10];
  const float* BW   = (const float*)d_in[11];
  const float* Wo_h = (const float*)d_in[12];
  const float* bo_h = (const float*)d_in[13];
  const float* Wo_e = (const float*)d_in[14];
  const float* bo_e = (const float*)d_in[15];
  const float* ln1h_g = (const float*)d_in[16];
  const float* ln1h_b = (const float*)d_in[17];
  const float* ln1e_g = (const float*)d_in[18];
  const float* ln1e_b = (const float*)d_in[19];
  const float* W1   = (const float*)d_in[20];
  const float* b1   = (const float*)d_in[21];
  const float* W2   = (const float*)d_in[22];
  const float* b2   = (const float*)d_in[23];
  const float* ln2h_g = (const float*)d_in[24];
  const float* ln2h_b = (const float*)d_in[25];
  const int*   ei   = (const int*)d_in[26];

  const int N = in_sizes[0] / HID;
  const int E = in_sizes[26] / 2;

  float* outh = (float*)d_out;
  float* oute = outh + (size_t)N * HID;

  // ---- workspace layout ----
  char* p = (char*)d_ws;
  auto allocF = [&](size_t n) { float* r = (float*)p; p += n * sizeof(float); return r; };
  float* Qh    = allocF((size_t)N * HID);
  float* Kh    = allocF((size_t)N * HID);
  float* Vh    = allocF((size_t)N * HID);
  float* bufE1 = allocF((size_t)E * HID);   // Ew -> cA (in place)
  float* bufE2 = allocF((size_t)E * HID);   // Eb -> Oe (reuse)
  float* score = allocF((size_t)E * NHEAD);
  float* denom = allocF((size_t)N * NHEAD);
  float* agg   = allocF((size_t)N * HID);   // -> On (in place)
  float* rowR  = allocF((size_t)N * HID);
  float* hmid  = allocF((size_t)N * 2 * HID);
  _Float16* w16 = (_Float16*)p;             // 786432 halfs of f16 weights
  _Float16* Wq16  = w16 + 0 * 65536;
  _Float16* Wk16  = w16 + 1 * 65536;
  _Float16* Wv16  = w16 + 2 * 65536;
  _Float16* Wew16 = w16 + 3 * 65536;
  _Float16* Web16 = w16 + 4 * 65536;
  _Float16* Weo16 = w16 + 5 * 65536;
  _Float16* Woh16 = w16 + 6 * 65536;
  _Float16* Woe16 = w16 + 7 * 65536;
  _Float16* W116  = w16 + 8 * 65536;             // 131072
  _Float16* W216  = w16 + 8 * 65536 + 131072;    // 131072

  // ---- zero accumulators ----
  hipMemsetAsync(denom, 0, (size_t)N * NHEAD * sizeof(float), stream);
  hipMemsetAsync(agg,   0, (size_t)N * HID   * sizeof(float), stream);
  hipMemsetAsync(rowR,  0, (size_t)N * HID   * sizeof(float), stream);

  // ---- convert weights to f16 ----
  const int WN = HID * HID;
  grit_cvt_f16<<<(WN + 255) / 256, 256, 0, stream>>>(Wq,  Wq16,  WN);
  grit_cvt_f16<<<(WN + 255) / 256, 256, 0, stream>>>(Wk,  Wk16,  WN);
  grit_cvt_f16<<<(WN + 255) / 256, 256, 0, stream>>>(Wv,  Wv16,  WN);
  grit_cvt_f16<<<(WN + 255) / 256, 256, 0, stream>>>(Wew, Wew16, WN);
  grit_cvt_f16<<<(WN + 255) / 256, 256, 0, stream>>>(Web, Web16, WN);
  grit_cvt_f16<<<(WN + 255) / 256, 256, 0, stream>>>(Weo, Weo16, WN);
  grit_cvt_f16<<<(WN + 255) / 256, 256, 0, stream>>>(Wo_h, Woh16, WN);
  grit_cvt_f16<<<(WN + 255) / 256, 256, 0, stream>>>(Wo_e, Woe16, WN);
  grit_cvt_f16<<<(2 * WN + 255) / 256, 256, 0, stream>>>(W1, W116, 2 * WN);
  grit_cvt_f16<<<(2 * WN + 255) / 256, 256, 0, stream>>>(W2, W216, 2 * WN);

  const dim3 blk(256);
  dim3 gN((N + 63) / 64, HID / 64);
  dim3 gE((E + 63) / 64, HID / 64);

  // ---- node projections ----
  grit_gemm<<<gN, blk, 0, stream>>>(x, Wq16, nullptr, nullptr, Qh, N, HID, HID, 0);
  grit_gemm<<<gN, blk, 0, stream>>>(x, Wk16, nullptr, nullptr, Kh, N, HID, HID, 0);
  grit_gemm<<<gN, blk, 0, stream>>>(x, Wv16, nullptr, nullptr, Vh, N, HID, HID, 0);

  // ---- edge projections ----
  grit_gemm<<<gE, blk, 0, stream>>>(conn, Wew16, nullptr, nullptr, bufE1, E, HID, HID, 0);
  grit_gemm<<<gE, blk, 0, stream>>>(conn, Web16, beb,     nullptr, bufE2, E, HID, HID, GF_BIAS);

  // ---- edge message: cA = relu(ssqrt((Qh[dst]+Kh[src])*Ew) + Eb) ----
  grit_edge_msg<<<E, blk, 0, stream>>>(Qh, Kh, bufE1, bufE2, ei, E, bufE1);

  // ---- Oe = cA @ Weo + beo ----
  grit_gemm<<<gE, blk, 0, stream>>>(bufE1, Weo16, beo, nullptr, bufE2, E, HID, HID, GF_BIAS);

  // ---- scores + softmax denominators ----
  grit_score<<<(E * NHEAD + 255) / 256, blk, 0, stream>>>(bufE2, Aw, ei, E, score, denom);

  // ---- scatter agg / rowR ----
  grit_scatter<<<E, blk, 0, stream>>>(score, denom, Vh, bufE2, ei, E, agg, rowR);

  // ---- On = agg + rowR @ BW (per head), in place into agg ----
  grit_bw<<<N, blk, 0, stream>>>(rowR, BW, agg);

  // ---- output projections + residual ----
  grit_gemm<<<gN, blk, 0, stream>>>(agg,   Woh16, bo_h, x,    outh, N, HID, HID, GF_BIAS | GF_RES);
  grit_gemm<<<gE, blk, 0, stream>>>(bufE2, Woe16, bo_e, conn, oute, E, HID, HID, GF_BIAS | GF_RES);

  // ---- LayerNorms (in place) ----
  grit_ln<<<N, blk, 0, stream>>>(outh, ln1h_g, ln1h_b);
  grit_ln<<<E, blk, 0, stream>>>(oute, ln1e_g, ln1e_b);

  // ---- FFN: hmid = relu(h1 @ W1 + b1); h = LN(h1 + hmid @ W2 + b2) ----
  dim3 gF1((N + 63) / 64, (2 * HID) / 64);
  grit_gemm<<<gF1, blk, 0, stream>>>(outh, W116, b1, nullptr, hmid, N, HID, 2 * HID, GF_BIAS | GF_RELU);
  grit_gemm<<<gN, blk, 0, stream>>>(hmid, W216, b2, outh, outh, N, 2 * HID, HID, GF_BIAS | GF_RES);
  grit_ln<<<N, blk, 0, stream>>>(outh, ln2h_g, ln2h_b);
}
